// CriticGraphPolicy_7456063226222
// MI455X (gfx1250) — compile-verified, matrix operand order checked by reading, not run on hardware
//
#include <hip/hip_runtime.h>
#include <hip/hip_bf16.h>
#include <math.h>

typedef __attribute__((ext_vector_type(16))) _Float16 v16h;
typedef __attribute__((ext_vector_type(8)))  _Float16 h8;
typedef __attribute__((ext_vector_type(8)))  float    v8f;

#define BATCH 8192
#define NUM_LIMBS 15

enum { ACT_NONE = 0, ACT_RELU = 1, ACT_TANH = 2 };

// ---------------------------------------------------------------------------
// Weight conversion: f32 [N,K] row-major -> f16 [Npad,Kpad] zero-padded,
// bias f32 [N] -> f32 [Npad] zero-padded.
// ---------------------------------------------------------------------------
__global__ __launch_bounds__(256) void k_cvt_w(const float* __restrict__ w,
                                               const float* __restrict__ b,
                                               _Float16* __restrict__ wo,
                                               float* __restrict__ bo,
                                               int N, int K, int Npad, int Kpad) {
  int tid = blockIdx.x * blockDim.x + threadIdx.x;
  int total = Npad * Kpad;
  if (tid >= total) return;
  int n = tid / Kpad, k = tid % Kpad;
  float v = (n < N && k < K) ? w[n * K + k] : 0.f;
  wo[tid] = (_Float16)v;
  if (k == 0) bo[n] = (n < N) ? b[n] : 0.f;
}

__global__ __launch_bounds__(256) void k_zero_f16(_Float16* __restrict__ p, int n) {
  int tid = blockIdx.x * blockDim.x + threadIdx.x;
  if (tid < n) p[tid] = (_Float16)0.f;
}

// ---------------------------------------------------------------------------
// Generic fused GEMM: C[M=8192, ldc] = act(A[M, lda(f16)] * W[N, ldw(f16)]^T + b)
// grid.x = 32 (M tiles of 256), grid.y = Npad64/64.
// Each wave: 32 rows x 64 cols -> acc[2][4] (64 VGPRs), 8 WMMAs per K-step.
// All fragment loads for a K-step are issued before the WMMA block so the
// backend can clause them and overlap the next step's loads with current math.
//
// 16-bit A 16x32 fragment: lane l -> row (l&15); half (l>>4) selects K block:
// v16h elems 0..7 = K k0+half*8..+7, elems 8..15 = K k0+16+half*8..+7.
// B mirrors A with lane -> n (W row-major [N,K] == B column-major).
// C/D: lane l -> col (l&15); VGPR v -> row v + 8*(l>>4).
// ---------------------------------------------------------------------------
template <int ACT, int OUTF32>
__global__ __launch_bounds__(256) void k_gemm_wmma(
    const _Float16* __restrict__ A, int lda,
    const _Float16* __restrict__ W, int ldw,
    const float* __restrict__ bias,
    void* __restrict__ Cout, int ldc,
    int N_store, int Kpad) {
  const int lane = threadIdx.x & 31;
  const int wave = threadIdx.x >> 5;
  const int mr   = lane & 15;
  const int half = lane >> 4;
  const int m_base = blockIdx.x * 256 + wave * 32;
  const int n_base = blockIdx.y * 64;

  v8f acc[2][4];
#pragma unroll
  for (int s = 0; s < 2; ++s)
#pragma unroll
    for (int t = 0; t < 4; ++t)
      acc[s][t] = (v8f){0.f, 0.f, 0.f, 0.f, 0.f, 0.f, 0.f, 0.f};

  const _Float16* ap0 = A + (size_t)(m_base + mr) * lda + half * 8;
  const _Float16* ap1 = ap0 + (size_t)16 * lda;
  const _Float16* bp0 = W + (size_t)(n_base + mr) * ldw + half * 8;

  for (int k0 = 0; k0 < Kpad; k0 += 32) {
    // ---- issue all loads for this K-step first ----
    h8 a0lo = *(const h8*)(ap0 + k0);
    h8 a0hi = *(const h8*)(ap0 + k0 + 16);
    h8 a1lo = *(const h8*)(ap1 + k0);
    h8 a1hi = *(const h8*)(ap1 + k0 + 16);
    h8 blo[4], bhi[4];
#pragma unroll
    for (int t = 0; t < 4; ++t) {
      const _Float16* bp = bp0 + (size_t)(t * 16) * ldw + k0;
      blo[t] = *(const h8*)(bp);
      bhi[t] = *(const h8*)(bp + 16);
    }
    v16h a0 = __builtin_shufflevector(a0lo, a0hi, 0, 1, 2, 3, 4, 5, 6, 7,
                                      8, 9, 10, 11, 12, 13, 14, 15);
    v16h a1 = __builtin_shufflevector(a1lo, a1hi, 0, 1, 2, 3, 4, 5, 6, 7,
                                      8, 9, 10, 11, 12, 13, 14, 15);
    // ---- 8 independent WMMAs ----
#pragma unroll
    for (int t = 0; t < 4; ++t) {
      v16h bv = __builtin_shufflevector(blo[t], bhi[t], 0, 1, 2, 3, 4, 5, 6, 7,
                                        8, 9, 10, 11, 12, 13, 14, 15);
      acc[0][t] = __builtin_amdgcn_wmma_f32_16x16x32_f16(
          false, a0, false, bv, (short)0, acc[0][t], false, false);
      acc[1][t] = __builtin_amdgcn_wmma_f32_16x16x32_f16(
          false, a1, false, bv, (short)0, acc[1][t], false, false);
    }
  }

#pragma unroll
  for (int s = 0; s < 2; ++s) {
#pragma unroll
    for (int t = 0; t < 4; ++t) {
      int col = n_base + t * 16 + mr;
      if (col < N_store) {
        float bc = bias[col];
#pragma unroll
        for (int v = 0; v < 8; ++v) {
          int row = m_base + s * 16 + v + 8 * half;
          float val = acc[s][t][v] + bc;
          if (ACT == ACT_RELU) val = fmaxf(val, 0.f);
          if (ACT == ACT_TANH) val = tanhf(val);
          if (OUTF32)
            ((float*)Cout)[(size_t)row * ldc + col] = val;
          else
            ((_Float16*)Cout)[(size_t)row * ldc + col] = (_Float16)val;
        }
      }
    }
  }
}

// ---------------------------------------------------------------------------
// Wave helpers (wave32)
// ---------------------------------------------------------------------------
__device__ __forceinline__ float wave_sum(float v) {
#pragma unroll
  for (int off = 16; off >= 1; off >>= 1) v += __shfl_xor(v, off, 32);
  return v;
}

// ---------------------------------------------------------------------------
// Pack / normalize / scatter kernels
// ---------------------------------------------------------------------------
// out[r,0:32]=x[node], out[r,32]=u[node], rest 0. (fc1 input, Kpad=64)
__global__ __launch_bounds__(256) void k_pack_xu(const float* __restrict__ x,
                                                 const float* __restrict__ u,
                                                 _Float16* __restrict__ out, int node) {
  int tid = blockIdx.x * blockDim.x + threadIdx.x;  // 8192*64
  int r = tid >> 6, c = tid & 63;
  float v = 0.f;
  if (c < 32)       v = x[((size_t)node * BATCH + r) * 32 + c];
  else if (c == 32) v = u[(size_t)node * BATCH + r];
  out[tid] = (_Float16)v;
}

// normalize 64-wide f32 rows, apply tanh, write into h[:,0:64] (ld 128)
__global__ __launch_bounds__(256) void k_norm_tanh64(const float* __restrict__ in,
                                                     _Float16* __restrict__ h) {
  int lane = threadIdx.x & 31, wave = threadIdx.x >> 5;
  int row = blockIdx.x * 8 + wave;
  float v0 = in[(size_t)row * 64 + lane];
  float v1 = in[(size_t)row * 64 + 32 + lane];
  float s = wave_sum(v0 * v0 + v1 * v1);
  float inv = 1.f / fmaxf(sqrtf(s), 1e-12f);
  h[(size_t)row * 128 + lane]      = (_Float16)tanhf(v0 * inv);
  h[(size_t)row * 128 + 32 + lane] = (_Float16)tanhf(v1 * inv);
}

// h[:,64:128] = tanh(concat(msg_up[c1], msg_up[c2])) or 0 for leaves
__global__ __launch_bounds__(256) void k_pack_children_tanh(
    const _Float16* __restrict__ msg_up, _Float16* __restrict__ h,
    int c1, int c2, int has) {
  int tid = blockIdx.x * blockDim.x + threadIdx.x;  // 8192*64
  int r = tid >> 6, c = tid & 63;
  float v = 0.f;
  if (has) {
    if (c < 32) v = (float)msg_up[((size_t)c1 * BATCH + r) * 32 + c];
    else        v = (float)msg_up[((size_t)c2 * BATCH + r) * 32 + (c - 32)];
  }
  h[(size_t)r * 128 + 64 + c] = (_Float16)tanhf(v);
}

// normalize 32-wide f32 rows -> f16 msg_up[node]
__global__ __launch_bounds__(256) void k_norm32(const float* __restrict__ in,
                                                _Float16* __restrict__ out) {
  int lane = threadIdx.x & 31, wave = threadIdx.x >> 5;
  int row = blockIdx.x * 8 + wave;
  float v = in[(size_t)row * 32 + lane];
  float s = wave_sum(v * v);
  float inv = 1.f / fmaxf(sqrtf(s), 1e-12f);
  out[(size_t)row * 32 + lane] = (_Float16)(v * inv);
}

// xum[r,0:32]=msg_up, [32]=u, [33:65]=msg_in, rest 0 (Kpad=96)
__global__ __launch_bounds__(256) void k_pack_xum(const _Float16* __restrict__ mu,
                                                  const float* __restrict__ u_node,
                                                  const _Float16* __restrict__ mi,
                                                  _Float16* __restrict__ out) {
  int tid = blockIdx.x * blockDim.x + threadIdx.x;  // 8192*96
  int r = tid / 96, c = tid % 96;
  float v = 0.f;
  if (c < 32)       v = (float)mu[(size_t)r * 32 + c];
  else if (c == 32) v = u_node[r];
  else if (c < 65)  v = (float)mi[(size_t)r * 32 + (c - 33)];
  out[tid] = (_Float16)v;
}

// xm = tanh(concat(msg_up, msg_in)) (Kpad=64)
__global__ __launch_bounds__(256) void k_pack_xm(const _Float16* __restrict__ mu,
                                                 const _Float16* __restrict__ mi,
                                                 _Float16* __restrict__ out) {
  int tid = blockIdx.x * blockDim.x + threadIdx.x;  // 8192*64
  int r = tid >> 6, c = tid & 63;
  float v = (c < 32) ? (float)mu[(size_t)r * 32 + c]
                     : (float)mi[(size_t)r * 32 + (c - 32)];
  out[tid] = (_Float16)tanhf(v);
}

// normalize 64-wide f32 rows, scatter halves to children's msg_in
__global__ __launch_bounds__(256) void k_norm64_scatter(const float* __restrict__ in,
                                                        _Float16* __restrict__ msg_in,
                                                        int c1, int c2) {
  int lane = threadIdx.x & 31, wave = threadIdx.x >> 5;
  int row = blockIdx.x * 8 + wave;
  float v0 = in[(size_t)row * 64 + lane];
  float v1 = in[(size_t)row * 64 + 32 + lane];
  float s = wave_sum(v0 * v0 + v1 * v1);
  float inv = 1.f / fmaxf(sqrtf(s), 1e-12f);
  msg_in[((size_t)c1 * BATCH + row) * 32 + lane] = (_Float16)(v0 * inv);
  msg_in[((size_t)c2 * BATCH + row) * 32 + lane] = (_Float16)(v1 * inv);
}

// final 300->1 layer, f32 weights straight from d_in; writes q[node, row, qi]
__global__ __launch_bounds__(256) void k_gemv300(const _Float16* __restrict__ in,
                                                 const float* __restrict__ w,
                                                 const float* __restrict__ b,
                                                 float* __restrict__ out,
                                                 int node, int qi) {
  int lane = threadIdx.x & 31, wave = threadIdx.x >> 5;
  int row = blockIdx.x * 8 + wave;
  float s = 0.f;
  for (int k = lane; k < 300; k += 32)
    s += (float)in[(size_t)row * 320 + k] * w[k];
  s = wave_sum(s);
  if (lane == 0)
    out[((size_t)node * BATCH + row) * 2 + qi] = s + b[0];
}

// ---------------------------------------------------------------------------
extern "C" void kernel_launch(void* const* d_in, const int* in_sizes, int n_in,
                              void* d_out, int out_size, void* d_ws, size_t ws_size,
                              hipStream_t stream) {
  (void)in_sizes; (void)n_in; (void)out_size; (void)ws_size;
  const float* x     = (const float*)d_in[0];
  const float* u     = (const float*)d_in[1];
  const float* up1w  = (const float*)d_in[2];  const float* up1b = (const float*)d_in[3];
  const float* up2w  = (const float*)d_in[4];  const float* up2b = (const float*)d_in[5];
  const float* up3w  = (const float*)d_in[6];  const float* up3b = (const float*)d_in[7];
  const float* q1l1w = (const float*)d_in[8];  const float* q1l1b = (const float*)d_in[9];
  const float* q1l2w = (const float*)d_in[10]; const float* q1l2b = (const float*)d_in[11];
  const float* q1l3w = (const float*)d_in[12]; const float* q1l3b = (const float*)d_in[13];
  const float* q2l1w = (const float*)d_in[14]; const float* q2l1b = (const float*)d_in[15];
  const float* q2l2w = (const float*)d_in[16]; const float* q2l2b = (const float*)d_in[17];
  const float* q2l3w = (const float*)d_in[18]; const float* q2l3b = (const float*)d_in[19];
  const float* mbl1w = (const float*)d_in[20]; const float* mbl1b = (const float*)d_in[21];
  const float* mbl2w = (const float*)d_in[22]; const float* mbl2b = (const float*)d_in[23];
  const float* mbl3w = (const float*)d_in[24]; const float* mbl3b = (const float*)d_in[25];
  float* qout = (float*)d_out;

  char* ws = (char*)d_ws;
  size_t off = 0;
  auto alloc = [&](size_t bytes) -> char* {
    char* p = ws + off;
    off = (off + bytes + 255) & ~(size_t)255;
    return p;
  };

  // f16 weights (padded)
  _Float16* w_fc1  = (_Float16*)alloc((size_t)64 * 64 * 2);
  _Float16* w_fc2  = (_Float16*)alloc((size_t)64 * 128 * 2);
  _Float16* w_fc3  = (_Float16*)alloc((size_t)64 * 64 * 2);
  _Float16* w_q1l1 = (_Float16*)alloc((size_t)448 * 96 * 2);
  _Float16* w_q1l2 = (_Float16*)alloc((size_t)320 * 416 * 2);
  _Float16* w_q2l1 = (_Float16*)alloc((size_t)448 * 96 * 2);
  _Float16* w_q2l2 = (_Float16*)alloc((size_t)320 * 416 * 2);
  _Float16* w_mbl1 = (_Float16*)alloc((size_t)448 * 64 * 2);
  _Float16* w_mbl2 = (_Float16*)alloc((size_t)320 * 416 * 2);
  _Float16* w_mbl3 = (_Float16*)alloc((size_t)64 * 320 * 2);
  // f32 biases (padded)
  float* b_fc1  = (float*)alloc(64 * 4);
  float* b_fc2  = (float*)alloc(64 * 4);
  float* b_fc3  = (float*)alloc(64 * 4);
  float* b_q1l1 = (float*)alloc(448 * 4);
  float* b_q1l2 = (float*)alloc(320 * 4);
  float* b_q2l1 = (float*)alloc(448 * 4);
  float* b_q2l2 = (float*)alloc(320 * 4);
  float* b_mbl1 = (float*)alloc(448 * 4);
  float* b_mbl2 = (float*)alloc(320 * 4);
  float* b_mbl3 = (float*)alloc(64 * 4);
  // activations
  _Float16* msg_up = (_Float16*)alloc((size_t)NUM_LIMBS * BATCH * 32 * 2);
  _Float16* msg_in = (_Float16*)alloc((size_t)NUM_LIMBS * BATCH * 32 * 2);
  _Float16* xu_in  = (_Float16*)alloc((size_t)BATCH * 64 * 2);
  float*    f32tmp = (float*)   alloc((size_t)BATCH * 64 * 4);
  _Float16* hbuf   = (_Float16*)alloc((size_t)BATCH * 128 * 2);
  _Float16* h2     = (_Float16*)alloc((size_t)BATCH * 64 * 2);
  _Float16* xum    = (_Float16*)alloc((size_t)BATCH * 96 * 2);
  _Float16* hid400 = (_Float16*)alloc((size_t)BATCH * 416 * 2);
  _Float16* hid300 = (_Float16*)alloc((size_t)BATCH * 320 * 2);
  _Float16* xm     = (_Float16*)alloc((size_t)BATCH * 64 * 2);

  auto cvt = [&](const float* w, const float* b, _Float16* wo, float* bo,
                 int N, int K, int Np, int Kp) {
    int total = Np * Kp;
    k_cvt_w<<<(total + 255) / 256, 256, 0, stream>>>(w, b, wo, bo, N, K, Np, Kp);
  };
  cvt(up1w, up1b, w_fc1, b_fc1, 64, 33, 64, 64);
  cvt(up2w, up2b, w_fc2, b_fc2, 64, 128, 64, 128);
  cvt(up3w, up3b, w_fc3, b_fc3, 32, 64, 64, 64);
  cvt(q1l1w, q1l1b, w_q1l1, b_q1l1, 400, 65, 448, 96);
  cvt(q1l2w, q1l2b, w_q1l2, b_q1l2, 300, 400, 320, 416);
  cvt(q2l1w, q2l1b, w_q2l1, b_q2l1, 400, 65, 448, 96);
  cvt(q2l2w, q2l2b, w_q2l2, b_q2l2, 300, 400, 320, 416);
  cvt(mbl1w, mbl1b, w_mbl1, b_mbl1, 400, 64, 448, 64);
  cvt(mbl2w, mbl2b, w_mbl2, b_mbl2, 300, 400, 320, 416);
  cvt(mbl3w, mbl3b, w_mbl3, b_mbl3, 64, 300, 64, 320);

  const int msg_elems = NUM_LIMBS * BATCH * 32;
  k_zero_f16<<<(msg_elems + 255) / 256, 256, 0, stream>>>(msg_up, msg_elems);
  k_zero_f16<<<(msg_elems + 255) / 256, 256, 0, stream>>>(msg_in, msg_elems);

  // ---- bottom-up: children (higher idx) processed before parents ----
  for (int node = NUM_LIMBS - 1; node >= 0; --node) {
    k_pack_xu<<<(BATCH * 64) / 256, 256, 0, stream>>>(x, u, xu_in, node);
    k_gemm_wmma<ACT_NONE, 1><<<dim3(32, 1), 256, 0, stream>>>(
        xu_in, 64, w_fc1, 64, b_fc1, f32tmp, 64, 64, 64);
    k_norm_tanh64<<<BATCH / 8, 256, 0, stream>>>(f32tmp, hbuf);
    int c1 = 2 * node + 1, c2 = 2 * node + 2;
    int has = (c2 < NUM_LIMBS) ? 1 : 0;
    k_pack_children_tanh<<<(BATCH * 64) / 256, 256, 0, stream>>>(msg_up, hbuf, c1, c2, has);
    k_gemm_wmma<ACT_TANH, 0><<<dim3(32, 1), 256, 0, stream>>>(
        hbuf, 128, w_fc2, 128, b_fc2, h2, 64, 64, 128);
    k_gemm_wmma<ACT_NONE, 1><<<dim3(32, 1), 256, 0, stream>>>(
        h2, 64, w_fc3, 64, b_fc3, f32tmp, 32, 32, 64);
    k_norm32<<<BATCH / 8, 256, 0, stream>>>(f32tmp, msg_up + (size_t)node * BATCH * 32);
  }

  // ---- top-down: parents before children ----
  for (int node = 0; node < NUM_LIMBS; ++node) {
    const _Float16* mu = msg_up + (size_t)node * BATCH * 32;
    const _Float16* mi = msg_in + (size_t)node * BATCH * 32;
    k_pack_xum<<<(BATCH * 96) / 256, 256, 0, stream>>>(mu, u + (size_t)node * BATCH, mi, xum);
    // Q1
    k_gemm_wmma<ACT_RELU, 0><<<dim3(32, 7), 256, 0, stream>>>(
        xum, 96, w_q1l1, 96, b_q1l1, hid400, 416, 416, 96);
    k_gemm_wmma<ACT_RELU, 0><<<dim3(32, 5), 256, 0, stream>>>(
        hid400, 416, w_q1l2, 416, b_q1l2, hid300, 320, 320, 416);
    k_gemv300<<<BATCH / 8, 256, 0, stream>>>(hid300, q1l3w, q1l3b, qout, node, 0);
    // Q2
    k_gemm_wmma<ACT_RELU, 0><<<dim3(32, 7), 256, 0, stream>>>(
        xum, 96, w_q2l1, 96, b_q2l1, hid400, 416, 416, 96);
    k_gemm_wmma<ACT_RELU, 0><<<dim3(32, 5), 256, 0, stream>>>(
        hid400, 416, w_q2l2, 416, b_q2l2, hid300, 320, 320, 416);
    k_gemv300<<<BATCH / 8, 256, 0, stream>>>(hid300, q2l3w, q2l3b, qout, node, 1);
    // child messages (only needed for non-leaf nodes)
    int c1 = 2 * node + 1, c2 = 2 * node + 2;
    if (c2 < NUM_LIMBS) {
      k_pack_xm<<<(BATCH * 64) / 256, 256, 0, stream>>>(mu, mi, xm);
      k_gemm_wmma<ACT_RELU, 0><<<dim3(32, 7), 256, 0, stream>>>(
          xm, 64, w_mbl1, 64, b_mbl1, hid400, 416, 416, 64);
      k_gemm_wmma<ACT_RELU, 0><<<dim3(32, 5), 256, 0, stream>>>(
          hid400, 416, w_mbl2, 416, b_mbl2, hid300, 320, 320, 416);
      k_gemm_wmma<ACT_NONE, 1><<<dim3(32, 1), 256, 0, stream>>>(
          hid300, 320, w_mbl3, 320, b_mbl3, f32tmp, 64, 64, 320);
      k_norm64_scatter<<<BATCH / 8, 256, 0, stream>>>(f32tmp, msg_in, c1, c2);
    }
  }
}
